// UnifiedMambaBlock_3521873182899
// MI455X (gfx1250) — matrix operand order repaired and checked by reading, not compiled, not run on hardware
//
#include <hip/hip_runtime.h>

// ---------------------------------------------------------------------------
// Mamba block on gfx1250: fully-unrolled, triple-buffered WMMA f16 GEMMs with
// sched_barrier-enforced load/WMMA pipelining + depthwise conv + lane-parallel
// selective scan.
// ---------------------------------------------------------------------------

typedef _Float16 half8 __attribute__((ext_vector_type(8)));
typedef _Float16 v16h  __attribute__((ext_vector_type(16)));
typedef float    v8f   __attribute__((ext_vector_type(8)));

#define D_MODEL 768
#define D_INNER 1536
#define DT_RANK 48
#define DT_PAD  64
#define D_STATE 16
#define NBATCH  2
#define SEQ     1024
#define MROWS   (NBATCH * SEQ)   // 2048
#define NXZ     (2 * D_INNER)    // 3072
#define NXPROJ  80               // 48 dt + 16 B + 16 C

// ----------------------------- conversions ---------------------------------

__global__ void f32_to_f16_kernel(const float* __restrict__ src,
                                  _Float16* __restrict__ dst, int n) {
  int i = blockIdx.x * blockDim.x + threadIdx.x;
  if (i < n) dst[i] = (_Float16)src[i];
}

// dst[r, c] = c < copy_cols ? src[r*src_stride + c] : 0   (f32 -> f16)
__global__ void pad_f32_to_f16_kernel(const float* __restrict__ src,
                                      _Float16* __restrict__ dst,
                                      int rows, int src_stride,
                                      int copy_cols, int dst_cols) {
  int i = blockIdx.x * blockDim.x + threadIdx.x;
  if (i >= rows * dst_cols) return;
  int r = i / dst_cols;
  int c = i - r * dst_cols;
  dst[i] = (c < copy_cols) ? (_Float16)src[(size_t)r * src_stride + c]
                           : (_Float16)0.0f;
}

// ----------------------------- WMMA GEMM -----------------------------------
// Out[m,n] = sum_k A[m,k] * W[n,k]   (A: [M,K] f16, W: [N,K] f16, Out f32)
// Each wave computes a 16 x (16*NT) tile; 8 waves per block stack along M.
// K is compile-time; chunk loop fully unrolled (pure SSA, no copies).
// Triple-buffered with __builtin_amdgcn_sched_barrier(0) fences so the
// machine scheduler cannot sink the chunk-(c+2) loads next to their WMMAs:
// order is pinned to [loads c+2][wmma c][loads c+3][wmma c+1]... giving two
// chunks of loads in flight behind every WMMA group (partial waits, not 0).
// EPI == 1: Out = softplus(acc + bias[n])  (dt_proj epilogue)

template <int NT, int EPI, int K>
__global__ __launch_bounds__(256, 1) void wmma_gemm_kernel(
    const _Float16* __restrict__ A, const _Float16* __restrict__ W,
    float* __restrict__ Out, int M, int N,
    const float* __restrict__ bias) {
  const int lane = threadIdx.x & 31;
  const int wave = threadIdx.x >> 5;
  const int g    = lane >> 4;   // half-wave group (0/1)
  const int l16  = lane & 15;
  const int m0   = (blockIdx.y * 8 + wave) * 16;
  const int n0   = blockIdx.x * (16 * NT);

  const _Float16* arow = A + (size_t)(m0 + l16) * K;
  const _Float16* wrow[NT];
#pragma unroll
  for (int t = 0; t < NT; ++t)
    wrow[t] = W + (size_t)(n0 + t * 16 + l16) * K;

  // A fragment, 16-bit 16x32 layout:
  //   e=0..7  : K = kb + g*8 + e      e=8..15 : K = kb + 16 + g*8 + (e-8)
  auto loadA = [&](int kb) {
    v16h af;
    half8 lo = *(const half8*)(arow + kb + g * 8);
    half8 hi = *(const half8*)(arow + kb + 16 + g * 8);
#pragma unroll
    for (int i = 0; i < 8; ++i) { af[i] = lo[i]; af[i + 8] = hi[i]; }
    return af;
  };
  // B fragment (32x16, col N = l16): same K order as A (e<8: kb+g*8+e, e>=8: kb+16+g*8+(e-8)).
  auto loadB = [&](int t, int kb) {
    v16h bf;
    half8 lo = *(const half8*)(wrow[t] + kb + g * 8);
    half8 hi = *(const half8*)(wrow[t] + kb + 16 + g * 8);
#pragma unroll
    for (int i = 0; i < 8; ++i) { bf[i] = lo[i]; bf[i + 8] = hi[i]; }
    return bf;
  };

  v8f acc[NT];
#pragma unroll
  for (int t = 0; t < NT; ++t) acc[t] = (v8f)(0.0f);

  constexpr int NCH = K / 32;                      // 2 / 24 / 48 here
  constexpr int LA  = (NCH >= 3) ? 2 : (NCH - 1);  // lookahead depth

  v16h ab[3];
  v16h bb[3][NT];
#pragma unroll
  for (int c = 0; c < LA; ++c) {                   // prime LA chunks
    ab[c] = loadA(c * 32);
#pragma unroll
    for (int t = 0; t < NT; ++t) bb[c][t] = loadB(t, c * 32);
  }
  __builtin_amdgcn_sched_barrier(0);

#pragma unroll
  for (int c = 0; c < NCH; ++c) {
    const int slot = c % 3;
    const int pc   = c + LA;                       // chunk to prefetch
    if (pc < NCH) {
      const int ps = pc % 3;
      ab[ps] = loadA(pc * 32);
#pragma unroll
      for (int t = 0; t < NT; ++t) bb[ps][t] = loadB(t, pc * 32);
    }
    __builtin_amdgcn_sched_barrier(0);             // loads stay above wmmas
#pragma unroll
    for (int t = 0; t < NT; ++t)
      acc[t] = __builtin_amdgcn_wmma_f32_16x16x32_f16(
          false, ab[slot], false, bb[slot][t], (short)0, acc[t], false, false);
    __builtin_amdgcn_sched_barrier(0);             // wmmas stay above next loads
  }

  // C/D layout: VGPR r, lane L -> row = r + 8*(L/16), col = L%16
#pragma unroll
  for (int t = 0; t < NT; ++t) {
    int col = n0 + t * 16 + l16;
#pragma unroll
    for (int r = 0; r < 8; ++r) {
      int row = m0 + g * 8 + r;
      float v = acc[t][r];
      if (EPI == 1) {
        v += bias[col];
        v = (v > 20.0f) ? v : __logf(1.0f + __expf(v));  // softplus
      }
      Out[(size_t)row * N + col] = v;
    }
  }
}

// ------------------------ depthwise conv + SiLU ----------------------------
// x = xz[:, 0:D_INNER]; x_conv = silu(causal_conv4(x) + bias)

__global__ void conv_silu_kernel(const float* __restrict__ xz,
                                 const float* __restrict__ cw,   // [D_INNER,4]
                                 const float* __restrict__ cb,   // [D_INNER]
                                 float* __restrict__ xconv,
                                 _Float16* __restrict__ xconv16) {
  int idx = blockIdx.x * blockDim.x + threadIdx.x;
  if (idx >= MROWS * D_INNER) return;
  int bs = idx / D_INNER;
  int d  = idx - bs * D_INNER;
  int s  = bs & (SEQ - 1);
  float acc = cb[d];
#pragma unroll
  for (int j = 0; j < 4; ++j) {
    int ss = s - 3 + j;
    if (ss >= 0) acc += cw[d * 4 + j] * xz[(size_t)(bs - 3 + j) * NXZ + d];
  }
  float v = acc / (1.0f + __expf(-acc));  // silu
  xconv[idx]   = v;
  xconv16[idx] = (_Float16)v;
}

// --------------------------- selective scan --------------------------------
// One LANE per (b, d, n) state element. Each wave owns two channels:
//   lanes 0..15  -> channel 2w,   n = lane
//   lanes 16..31 -> channel 2w+1, n = lane-16
// Per timestep: 1 exp per lane; <h,C> via 4-step shfl_xor butterfly that
// never crosses the 16-lane half-wave boundary (masks 1,2,4,8).

__global__ __launch_bounds__(256) void scan_kernel(
    const float* __restrict__ dt,      // [MROWS, D_INNER]
    const float* __restrict__ xconv,   // [MROWS, D_INNER]
    const float* __restrict__ dtBC,    // [MROWS, 80] (B at +48, C at +64)
    const float* __restrict__ xz,      // [MROWS, 3072] (z at +D_INNER)
    const float* __restrict__ A_log,   // [D_INNER, 16]
    const float* __restrict__ D_param, // [D_INNER]
    _Float16* __restrict__ y16)        // [MROWS, D_INNER]
{
  const int lane = threadIdx.x & 31;
  const int wv   = (blockIdx.x * blockDim.x + threadIdx.x) >> 5;  // 0..1535
  const int n    = lane & 15;
  const int ch   = 2 * wv + (lane >> 4);  // 0..3071
  const int b    = ch / D_INNER;
  const int d    = ch - b * D_INNER;

  const float An = -__expf(A_log[d * D_STATE + n]);
  const float Dp = D_param[d];

  float h = 0.0f;
  for (int s = 0; s < SEQ; ++s) {
    int bs = b * SEQ + s;
    float dtv = dt[(size_t)bs * D_INNER + d];      // uniform across 16 lanes
    float xc  = xconv[(size_t)bs * D_INNER + d];   // uniform across 16 lanes
    float Bm  = dtBC[(size_t)bs * NXPROJ + DT_RANK + n];           // coalesced
    float Cm  = dtBC[(size_t)bs * NXPROJ + DT_RANK + D_STATE + n]; // coalesced

    if (s + 1 < SEQ) {  // prefetch next timestep's operands
      __builtin_prefetch(&dt[(size_t)(bs + 1) * D_INNER + d], 0, 3);
      __builtin_prefetch(&dtBC[(size_t)(bs + 1) * NXPROJ + DT_RANK], 0, 3);
    }

    h = h * __expf(dtv * An) + (dtv * xc) * Bm;
    float p = h * Cm;
    p += __shfl_xor(p, 1);
    p += __shfl_xor(p, 2);
    p += __shfl_xor(p, 4);
    p += __shfl_xor(p, 8);   // all 16 lanes of the half now hold <h,C>

    float y  = p + Dp * xc;
    float z  = xz[(size_t)bs * NXZ + D_INNER + d];
    float sz = z / (1.0f + __expf(-z));            // silu(z)
    if (n == 0) y16[(size_t)bs * D_INNER + d] = (_Float16)(y * sz);
  }
}

// ------------------------------- host --------------------------------------

extern "C" void kernel_launch(void* const* d_in, const int* in_sizes, int n_in,
                              void* d_out, int out_size, void* d_ws,
                              size_t ws_size, hipStream_t stream) {
  const float* u          = (const float*)d_in[0];   // [2,1024,768]
  const float* in_proj_w  = (const float*)d_in[1];   // [3072,768]
  const float* conv_w     = (const float*)d_in[2];   // [1536,1,4]
  const float* conv_b     = (const float*)d_in[3];   // [1536]
  const float* xp_dt_w    = (const float*)d_in[4];   // [48,1536]
  const float* xp_b_w     = (const float*)d_in[5];   // [16,1536]
  const float* xp_c_w     = (const float*)d_in[6];   // [16,1536]
  const float* dt_proj_w  = (const float*)d_in[7];   // [1536,48]
  const float* dt_proj_b  = (const float*)d_in[8];   // [1536]
  const float* A_log      = (const float*)d_in[9];   // [1536,16]
  const float* D_param    = (const float*)d_in[10];  // [1536]
  const float* out_proj_w = (const float*)d_in[11];  // [768,1536]
  float* out = (float*)d_out;                        // [2,1024,768]

  // Workspace carving (256B aligned slabs)
  char* wsp = (char*)d_ws;
  auto carve = [&](size_t bytes) {
    char* p = wsp;
    wsp += (bytes + 255) & ~(size_t)255;
    return p;
  };
  _Float16* u16    = (_Float16*)carve((size_t)MROWS * D_MODEL * 2);
  _Float16* w_in16 = (_Float16*)carve((size_t)NXZ * D_MODEL * 2);
  _Float16* w_xp16 = (_Float16*)carve((size_t)NXPROJ * D_INNER * 2);
  _Float16* w_dt16 = (_Float16*)carve((size_t)D_INNER * DT_PAD * 2);
  _Float16* w_out16= (_Float16*)carve((size_t)D_MODEL * D_INNER * 2);
  _Float16* dtr16  = (_Float16*)carve((size_t)MROWS * DT_PAD * 2);
  _Float16* xconv16= (_Float16*)carve((size_t)MROWS * D_INNER * 2);
  _Float16* y16    = (_Float16*)carve((size_t)MROWS * D_INNER * 2);
  float* xz    = (float*)carve((size_t)MROWS * NXZ * 4);
  float* xconv = (float*)carve((size_t)MROWS * D_INNER * 4);
  float* dtBC  = (float*)carve((size_t)MROWS * NXPROJ * 4);
  float* dtb   = (float*)carve((size_t)MROWS * D_INNER * 4);
  (void)ws_size; (void)n_in; (void)in_sizes; (void)out_size;

  const int BLK = 256;
  auto cdiv = [](int a, int b) { return (a + b - 1) / b; };

  // --- f32 -> f16 weight/activation conversions ---
  {
    int n = MROWS * D_MODEL;
    f32_to_f16_kernel<<<cdiv(n, BLK), BLK, 0, stream>>>(u, u16, n);
  }
  {
    int n = NXZ * D_MODEL;
    f32_to_f16_kernel<<<cdiv(n, BLK), BLK, 0, stream>>>(in_proj_w, w_in16, n);
  }
  {
    int n = D_MODEL * D_INNER;
    f32_to_f16_kernel<<<cdiv(n, BLK), BLK, 0, stream>>>(out_proj_w, w_out16, n);
  }
  {
    int n = DT_RANK * D_INNER;
    f32_to_f16_kernel<<<cdiv(n, BLK), BLK, 0, stream>>>(xp_dt_w, w_xp16, n);
    int m = D_STATE * D_INNER;
    f32_to_f16_kernel<<<cdiv(m, BLK), BLK, 0, stream>>>(
        xp_b_w, w_xp16 + (size_t)DT_RANK * D_INNER, m);
    f32_to_f16_kernel<<<cdiv(m, BLK), BLK, 0, stream>>>(
        xp_c_w, w_xp16 + (size_t)(DT_RANK + D_STATE) * D_INNER, m);
  }
  {
    int n = D_INNER * DT_PAD;
    pad_f32_to_f16_kernel<<<cdiv(n, BLK), BLK, 0, stream>>>(
        dt_proj_w, w_dt16, D_INNER, DT_RANK, DT_RANK, DT_PAD);
  }

  // --- GEMM1: xz = u @ in_proj_w^T   [2048 x 3072 x 768] ---
  {
    dim3 grid(NXZ / 64, MROWS / 128);
    wmma_gemm_kernel<4, 0, D_MODEL><<<grid, BLK, 0, stream>>>(
        u16, w_in16, xz, MROWS, NXZ, nullptr);
  }

  // --- causal depthwise conv + silu ---
  {
    int n = MROWS * D_INNER;
    conv_silu_kernel<<<cdiv(n, BLK), BLK, 0, stream>>>(xz, conv_w, conv_b,
                                                       xconv, xconv16);
  }

  // --- GEMM2: dtBC = x_conv @ [xp_dt;xp_b;xp_c]^T  [2048 x 80 x 1536] ---
  {
    dim3 grid(1, MROWS / 128);  // NT=5 -> all 80 cols per wave
    wmma_gemm_kernel<5, 0, D_INNER><<<grid, BLK, 0, stream>>>(
        xconv16, w_xp16, dtBC, MROWS, NXPROJ, nullptr);
  }

  // --- pad dt_r (first 48 cols of dtBC) to K=64 f16 ---
  {
    int n = MROWS * DT_PAD;
    pad_f32_to_f16_kernel<<<cdiv(n, BLK), BLK, 0, stream>>>(
        dtBC, dtr16, MROWS, NXPROJ, DT_RANK, DT_PAD);
  }

  // --- GEMM3: dt = softplus(dt_r @ dt_proj_w^T + b)  [2048 x 1536 x 64] ---
  {
    dim3 grid(D_INNER / 64, MROWS / 128);
    wmma_gemm_kernel<4, 1, DT_PAD><<<grid, BLK, 0, stream>>>(
        dtr16, w_dt16, dtb, MROWS, D_INNER, dt_proj_b);
  }

  // --- lane-parallel selective scan (+ D-skip + silu(z) gate) -> f16 y ---
  {
    int total_threads = (NBATCH * D_INNER / 2) * 32;  // 1536 waves
    scan_kernel<<<total_threads / BLK, BLK, 0, stream>>>(
        dtb, xconv, dtBC, xz, A_log, D_param, y16);
  }

  // --- GEMM4: out = y @ out_proj_w^T  [2048 x 768 x 1536] ---
  {
    dim3 grid(D_MODEL / 64, MROWS / 128);
    wmma_gemm_kernel<4, 0, D_INNER><<<grid, BLK, 0, stream>>>(
        y16, w_out16, out, MROWS, D_MODEL, nullptr);
  }
}
